// LatticeCrf_8272107012561
// MI455X (gfx1250) — matrix operand order repaired and checked by reading, not compiled, not run on hardware
//
#include <hip/hip_runtime.h>
#include <hip/hip_fp16.h>

typedef __attribute__((ext_vector_type(16))) _Float16 v16h;
typedef __attribute__((ext_vector_type(8)))  _Float16 v8h;
typedef __attribute__((ext_vector_type(8)))  float    v8f;
typedef __attribute__((ext_vector_type(4)))  float    v4f;

#define NB     32
#define NTT    2048
#define NV     32
#define NCHUNK 64
#define CLEN   (NTT / NCHUNK)   // 32 time steps per chunk

// ---------------------------------------------------------------------------
// Kernel 1: one wave (32 lanes) per (batch, chunk). Computes the log-semiring
// product of CLEN consecutive 32x32 matrices M_t = scores_t + transition,
// carried in the exp domain with per-row rescaling so every step is a real
// f32-accumulate matmul on the WMMA units:  Acc <- Acc x exp(M_t).
// The 256MB `inputs` stream is double-buffered with NT-hinted b128 loads.
// Result written to ws in log domain: ws[b][c][i][j].
// ---------------------------------------------------------------------------
__global__ __launch_bounds__(32)
void crf_chunk_kernel(const float* __restrict__ inputs,
                      const float* __restrict__ transition,
                      float* __restrict__ ws) {
  const int lane = threadIdx.x;            // 0..31
  const int b    = blockIdx.x / NCHUNK;
  const int c    = blockIdx.x % NCHUNK;
  const int t0   = c * CLEN;

  __shared__ alignas(16) _Float16 ldsPt[NV * NV];  // P transposed: Pt[j][i]
  __shared__ alignas(16) _Float16 ldsA [NV * NV];  // normalized Acc, row-major

  // b128 streaming layout: lane l, quarter q covers flat elements
  //   e = q*128 + l*4 + c  ->  row i = 4q + l/8, cols j = 4*(l%8)..+3
  const int ifrac = lane >> 3;        // l/8   (row sub-index)
  const int jbase = (lane & 7) * 4;   // first column of this lane's float4

  // matching transition tile (constant, kept in VGPRs)
  v4f tr[8];
  #pragma unroll
  for (int q = 0; q < 8; ++q)
    tr[q] = *(const v4f*)(transition + q * 128 + lane * 4);

  const int h = lane >> 4;   // lane half (C-layout M-group)
  const int n = lane & 15;   // column within 16-wide tile

  // Accumulator = identity (exp-domain log-semiring identity), 2x2 C tiles.
  // C/D layout: component k of acc[ti][tj] holds element
  //   (row = 16*ti + 8*h + k, col = 16*tj + n)
  v8f   acc[2][2];
  float lsc[2][8];                        // per-row log scale (replicated in half)
  #pragma unroll
  for (int ti = 0; ti < 2; ++ti) {
    #pragma unroll
    for (int k = 0; k < 8; ++k) lsc[ti][k] = 0.0f;
    #pragma unroll
    for (int tj = 0; tj < 2; ++tj)
      #pragma unroll
      for (int k = 0; k < 8; ++k)
        acc[ti][tj][k] = (ti == tj && (8 * h + k) == n) ? 1.0f : 0.0f;
  }

  const float* sbase = inputs + ((size_t)b * NTT + t0) * (NV * NV);

  // prologue: load step 0
  v4f sv[8];
  #pragma unroll
  for (int q = 0; q < 8; ++q)
    sv[q] = __builtin_nontemporal_load((const v4f*)(sbase + q * 128 + lane * 4));

  for (int t = 0; t < CLEN; ++t) {
    // ---- prefetch next step's 4KB matrix (double buffer) -----------------
    const float* smn = sbase + (size_t)((t + 1 < CLEN) ? t + 1 : t) * (NV * NV);
    v4f nxt[8];
    #pragma unroll
    for (int q = 0; q < 8; ++q)
      nxt[q] = __builtin_nontemporal_load((const v4f*)(smn + q * 128 + lane * 4));

    // ---- P = exp(scores + transition), stored transposed in LDS as f16 ---
    #pragma unroll
    for (int q = 0; q < 8; ++q) {
      const int i = 4 * q + ifrac;
      #pragma unroll
      for (int cc = 0; cc < 4; ++cc)
        ldsPt[(jbase + cc) * NV + i] = (_Float16)__expf(sv[q][cc] + tr[q][cc]);
    }

    // ---- renormalize Acc rows (max -> 1), stash f16 copy in LDS ----------
    #pragma unroll
    for (int ti = 0; ti < 2; ++ti) {
      #pragma unroll
      for (int k = 0; k < 8; ++k) {
        float rm = fmaxf(acc[ti][0][k], acc[ti][1][k]);
        #pragma unroll
        for (int m = 1; m < 16; m <<= 1)      // reduce within 16-lane half
          rm = fmaxf(rm, __shfl_xor(rm, m, 32));
        const float inv = __builtin_amdgcn_rcpf(rm);
        acc[ti][0][k] *= inv;
        acc[ti][1][k] *= inv;
        lsc[ti][k] += __logf(rm);
        const int row = 16 * ti + 8 * h + k;
        ldsA[row * NV +      n] = (_Float16)acc[ti][0][k];
        ldsA[row * NV + 16 + n] = (_Float16)acc[ti][1][k];
      }
    }
    __syncthreads();

    // ---- A fragments (16x32 f16): lane<16 -> row n, K {0..7,16..23};
    //                               lane>=16 -> row n, K {8..15,24..31} ----
    union AF { v16h v; v8h p[2]; };
    AF afrag[2];
    const int kofs = h ? 8 : 0;
    #pragma unroll
    for (int ti = 0; ti < 2; ++ti) {
      const _Float16* ar = &ldsA[(16 * ti + n) * NV];
      afrag[ti].p[0] = *(const v8h*)(ar + kofs);
      afrag[ti].p[1] = *(const v8h*)(ar + kofs + 16);
    }
    // ---- B fragments (32x16 f16): lane<16 -> N=n, K {0..7,16..23};
    //                               lane>=16 -> N=n, K {8..15,24..31} ------
    union BF { v16h v; v8h p[2]; };
    BF bfrag[2];
    #pragma unroll
    for (int tj = 0; tj < 2; ++tj) {
      const _Float16* br = &ldsPt[(16 * tj + n) * NV + 8 * h];
      bfrag[tj].p[0] = *(const v8h*)(br);
      bfrag[tj].p[1] = *(const v8h*)(br + 16);
    }
    __syncthreads();

    // ---- Acc' = Acc x P  (4x v_wmma_f32_16x16x32_f16, fresh C=0) ---------
    #pragma unroll
    for (int ti = 0; ti < 2; ++ti)
      #pragma unroll
      for (int tj = 0; tj < 2; ++tj) {
        v8f z = {};
        acc[ti][tj] = __builtin_amdgcn_wmma_f32_16x16x32_f16(
            false, afrag[ti].v, false, bfrag[tj].v, (short)0, z, false, false);
      }

    #pragma unroll
    for (int q = 0; q < 8; ++q) sv[q] = nxt[q];
  }

  // ---- emit chunk matrix back in log domain ------------------------------
  float* outp = ws + ((size_t)b * NCHUNK + c) * (NV * NV);
  #pragma unroll
  for (int ti = 0; ti < 2; ++ti)
    #pragma unroll
    for (int tj = 0; tj < 2; ++tj)
      #pragma unroll
      for (int k = 0; k < 8; ++k) {
        const int row = 16 * ti + 8 * h + k;
        const int col = 16 * tj + n;
        outp[row * NV + col] = __logf(acc[ti][tj][k]) + lsc[ti][k];
      }
}

// ---------------------------------------------------------------------------
// Kernel 2: one wave per batch. Folds the 64 chunk matrices into alpha_T with
// a stable logsumexp vec-mat scan, computes the denominator, the gathered
// numerator, and the final log-prob.
// ---------------------------------------------------------------------------
__global__ __launch_bounds__(32)
void crf_finish_kernel(const float* __restrict__ inputs,
                       const float* __restrict__ transition,
                       const float* __restrict__ initial,
                       const float* __restrict__ finalv,
                       const int*   __restrict__ targets,
                       const float* __restrict__ ws,
                       float* __restrict__ outv) {
  const int lane = threadIdx.x;
  const int b    = blockIdx.x;

  // ---- denominator: alpha scan over chunk matrices -----------------------
  float alpha = initial[lane];
  for (int c = 0; c < NCHUNK; ++c) {
    const float* W = ws + ((size_t)b * NCHUNK + c) * (NV * NV);
    float w[NV];
    #pragma unroll
    for (int i = 0; i < NV; ++i) w[i] = W[i * NV + lane];
    float m = -3.4e38f;
    #pragma unroll
    for (int i = 0; i < NV; ++i)
      m = fmaxf(m, __shfl(alpha, i, 32) + w[i]);
    float s = 0.0f;
    #pragma unroll
    for (int i = 0; i < NV; ++i)
      s += __expf(__shfl(alpha, i, 32) + w[i] - m);
    alpha = m + __logf(s);
  }
  float x = alpha + finalv[lane];
  float m = x;
  #pragma unroll
  for (int d = 1; d < 32; d <<= 1) m = fmaxf(m, __shfl_xor(m, d, 32));
  float e = __expf(x - m);
  #pragma unroll
  for (int d = 1; d < 32; d <<= 1) e += __shfl_xor(e, d, 32);
  const float denom = m + __logf(e);

  // ---- numerator: gathered path score ------------------------------------
  const int* tg = targets + (size_t)b * (NTT + 1);
  float num = 0.0f;
  for (int t = lane; t < NTT; t += 32) {
    const int yp = tg[t], yn = tg[t + 1];
    num += inputs[(((size_t)b * NTT + t) * NV + yp) * NV + yn]
         + transition[yp * NV + yn];
  }
  #pragma unroll
  for (int d = 1; d < 32; d <<= 1) num += __shfl_xor(num, d, 32);

  if (lane == 0)
    outv[b] = initial[tg[0]] + num + finalv[tg[NTT]] - denom;
}

// ---------------------------------------------------------------------------
extern "C" void kernel_launch(void* const* d_in, const int* in_sizes, int n_in,
                              void* d_out, int out_size, void* d_ws, size_t ws_size,
                              hipStream_t stream) {
  const float* inputs     = (const float*)d_in[0];
  const float* transition = (const float*)d_in[1];
  const float* initial    = (const float*)d_in[2];
  const float* finalv     = (const float*)d_in[3];
  const int*   targets    = (const int*)d_in[4];
  float*       ws         = (float*)d_ws;
  float*       outv       = (float*)d_out;

  crf_chunk_kernel<<<NB * NCHUNK, 32, 0, stream>>>(inputs, transition, ws);
  crf_finish_kernel<<<NB, 32, 0, stream>>>(inputs, transition, initial, finalv,
                                           targets, ws, outv);
}
